// LSTM_678604833060
// MI455X (gfx1250) — matrix operand order repaired and checked by reading, not compiled, not run on hardware
//
#include <hip/hip_runtime.h>

// LSTM: I=256, H=1024, O=1, T=16384
// Persistent-kernel design for MI455X (gfx1250):
//  - 64 workgroups (one per WGP), WG g owns hidden units [16g,16g+16) i.e.
//    gate rows {q*1024 + 16g + j : q<4, j<16} -> 64 rows.
//  - W_hh slice (64x1024) and W_ih slice (64x256) held in LDS as fp16 (loaded once).
//  - xg = x@W_ih.T + bias computed per 512-step chunk with v_wmma_f32_16x16x32_f16
//    into a private 512x64 f32 scratch region (no cross-WG dependency).
//  - Per step: async-load h (4KB, L2-resident, double-buffered) straight into LDS,
//    64x1024 GEMV from LDS, gates + cell update local, write 16 h values,
//    one 64-WG global barrier.
//  - Small head kernel computes relu(h@W1.T+b1)@W2.T+b2 -> scalar.

#define T_STEPS 16384
#define HID     1024
#define INDIM   256
#define NWG     64
#define NTHR    512
#define CHUNK   512
#define ROWS    64          // 4 gates * 16 hidden units per WG
#define WHH_STRIDE 1040     // 1024 + 16 pad (bank spread)
#define WIH_STRIDE 272      // 256 + 16 pad (row stride stays 32B-aligned)

typedef _Float16 f16;
typedef __attribute__((ext_vector_type(16))) _Float16 v16h;
typedef __attribute__((ext_vector_type(8)))  float    v8f;
typedef __attribute__((ext_vector_type(2)))  int      v2i;
typedef __attribute__((address_space(1))) v2i* gv2i_p;   // global int2*
typedef __attribute__((address_space(3))) v2i* lv2i_p;   // LDS int2*

#if __has_builtin(__builtin_amdgcn_global_load_async_to_lds_b64)
#define HAVE_ASYNC_LDS 1
#else
#define HAVE_ASYNC_LDS 0
#endif

__device__ __forceinline__ float sigmoidf_fast(float x) {
    return 1.f / (1.f + __expf(-x));
}

__global__ __launch_bounds__(NTHR, 1)
void lstm_persistent(const float* __restrict__ x,
                     const float* __restrict__ W_ih,
                     const float* __restrict__ W_hh,
                     const float* __restrict__ b_ih,
                     const float* __restrict__ b_hh,
                     unsigned*    __restrict__ cnt,    // T_STEPS counters (zeroed)
                     float*       __restrict__ hbuf,   // 2*HID f32 (zeroed)
                     float*       __restrict__ xg_ws)  // NWG*CHUNK*ROWS f32
{
    __shared__ f16   Whh_lds[ROWS * WHH_STRIDE];
    __shared__ f16   Wih_lds[ROWS * WIH_STRIDE];
    __shared__ float h_lds[HID];
    __shared__ float g_lds[ROWS];
    __shared__ float c_lds[16];
    __shared__ float bias_lds[ROWS];

    const int tid  = threadIdx.x;
    const int wg   = blockIdx.x;
    const int lane = tid & 31;
    const int wave = tid >> 5;

    // ---- stage weight slices into LDS as fp16 (once) ----
    for (int idx = tid; idx < ROWS * HID; idx += NTHR) {
        const int rl = idx >> 10, k = idx & (HID - 1);
        const int grow = (rl >> 4) * HID + wg * 16 + (rl & 15);
        Whh_lds[rl * WHH_STRIDE + k] = (f16)W_hh[(size_t)grow * HID + k];
    }
    for (int idx = tid; idx < ROWS * INDIM; idx += NTHR) {
        const int rl = idx >> 8, k = idx & (INDIM - 1);
        const int grow = (rl >> 4) * HID + wg * 16 + (rl & 15);
        Wih_lds[rl * WIH_STRIDE + k] = (f16)W_ih[(size_t)grow * INDIM + k];
    }
    if (tid < ROWS) {
        const int grow = (tid >> 4) * HID + wg * 16 + (tid & 15);
        bias_lds[tid] = b_ih[grow] + b_hh[grow];
    }
    if (tid < 16) c_lds[tid] = 0.f;
    __syncthreads();

    float* xg_my = xg_ws + (size_t)wg * (CHUNK * ROWS);

    for (int t = 0; t < T_STEPS; ++t) {
        // ================= chunk boundary: WMMA input projection =================
        if ((t & (CHUNK - 1)) == 0) {
            // A (16x32 f16): lane L<16 -> M=L, elems e -> K = e
            //                lane L>=16 -> M=L-16, elems e -> K = 16 + e
            // B (32x16 f16): lane l -> col N=l%16; elems e -> K = e + (l<16?0:16)
            // D (16x16 f32): VGPR i, lane L -> M = i + (L<16?0:8), N = L%16
            const int klo  = (lane < 16) ? 0 : 8;
            const int koff = (lane < 16) ? 0 : 16;
            const int mb   = (lane < 16) ? 0 : 8;
            const int ncol = lane & 15;

            // Load A for BOTH t-tiles of this wave up front (2 x 8 x v16h).
            // Each B tile is then consumed exactly once -> nothing to hoist/spill.
            v16h a[2][8];
            #pragma unroll
            for (int tt = 0; tt < 2; ++tt) {
                const int ttile = wave * 2 + tt;          // 0..31
                const int t0 = t + ttile * 16;
                const float* xrow = x + (size_t)(t0 + (lane & 15)) * INDIM;
                #pragma unroll
                for (int kk = 0; kk < 8; ++kk) {
                    const float4 f0 = *(const float4*)(xrow + kk * 32 + koff);
                    const float4 f1 = *(const float4*)(xrow + kk * 32 + koff + 4);
                    const float4 f2 = *(const float4*)(xrow + kk * 32 + koff + 8);
                    const float4 f3 = *(const float4*)(xrow + kk * 32 + koff + 12);
                    v16h av;
                    av[0] = (f16)f0.x; av[1] = (f16)f0.y; av[2]  = (f16)f0.z; av[3]  = (f16)f0.w;
                    av[4] = (f16)f1.x; av[5] = (f16)f1.y; av[6]  = (f16)f1.z; av[7]  = (f16)f1.w;
                    av[8] = (f16)f2.x; av[9] = (f16)f2.y; av[10] = (f16)f2.z; av[11] = (f16)f2.w;
                    av[12]= (f16)f3.x; av[13]= (f16)f3.y; av[14] = (f16)f3.z; av[15] = (f16)f3.w;
                    a[tt][kk] = av;
                }
            }

            for (int nt = 0; nt < 4; ++nt) {
                v8f acc0 = {0.f, 0.f, 0.f, 0.f, 0.f, 0.f, 0.f, 0.f};
                v8f acc1 = {0.f, 0.f, 0.f, 0.f, 0.f, 0.f, 0.f, 0.f};
                #pragma unroll
                for (int kk = 0; kk < 8; ++kk) {
                    // 32 contiguous bytes, 32B-aligned (row stride 544B, offs 64*kk+{0,32})
                    const v16h b = *(const v16h*)&Wih_lds[(nt * 16 + ncol) * WIH_STRIDE
                                                          + kk * 32 + koff];
                    acc0 = __builtin_amdgcn_wmma_f32_16x16x32_f16(
                               false, a[0][kk], false, b, (short)0, acc0, false, false);
                    acc1 = __builtin_amdgcn_wmma_f32_16x16x32_f16(
                               false, a[1][kk], false, b, (short)0, acc1, false, false);
                }
                const float bv = bias_lds[nt * 16 + ncol];
                float* dst0 = xg_my + (size_t)((wave * 2 + 0) * 16 + mb) * ROWS + nt * 16 + ncol;
                float* dst1 = xg_my + (size_t)((wave * 2 + 1) * 16 + mb) * ROWS + nt * 16 + ncol;
                #pragma unroll
                for (int i = 0; i < 8; ++i) {
                    dst0[(size_t)i * ROWS] = acc0[i] + bv;
                    dst1[(size_t)i * ROWS] = acc1[i] + bv;
                }
            }
            __threadfence();   // make xg visible across waves (L0 flush/inv)
            __syncthreads();
        }

        // ================= recurrent step =================
        // load h (double-buffered in L2 scratch) -> LDS
        const float* hsrc = hbuf + (t & 1) * HID;
#if HAVE_ASYNC_LDS
        // CDNA5 async global->LDS copy (ASYNCcnt): skips the VGPR round-trip.
        __builtin_amdgcn_global_load_async_to_lds_b64(
            (gv2i_p)(hsrc + tid * 2),
            (lv2i_p)&h_lds[tid * 2],
            0, 0);
  #if __has_builtin(__builtin_amdgcn_s_wait_asynccnt)
        __builtin_amdgcn_s_wait_asynccnt(0);
  #else
        asm volatile("s_wait_asynccnt 0" ::: "memory");
  #endif
#else
        ((float2*)h_lds)[tid] = ((const float2*)hsrc)[tid];
#endif
        __syncthreads();

        // 64 rows x 1024 GEMV from LDS: 8 threads per row
        const int rl = tid >> 3;
        const int q  = tid & 7;
        const f16*   wp = &Whh_lds[rl * WHH_STRIDE + q * 128];
        const float* hp = &h_lds[q * 128];
        float p = 0.f;
        #pragma unroll 8
        for (int k = 0; k < 128; ++k)
            p = __builtin_fmaf((float)wp[k], hp[k], p);
        p += __shfl_xor(p, 1, 32);
        p += __shfl_xor(p, 2, 32);
        p += __shfl_xor(p, 4, 32);
        if (q == 0)
            g_lds[rl] = p + xg_my[(size_t)(t & (CHUNK - 1)) * ROWS + rl];
        __syncthreads();

        // gates + cell update for this WG's 16 hidden units
        if (tid < 16) {
            const float gi = sigmoidf_fast(g_lds[tid]);
            const float gf = sigmoidf_fast(g_lds[16 + tid]);
            const float gg = tanhf(g_lds[32 + tid]);
            const float go = sigmoidf_fast(g_lds[48 + tid]);
            const float c  = gf * c_lds[tid] + gi * gg;
            c_lds[tid] = c;
            hbuf[((t + 1) & 1) * HID + wg * 16 + tid] = go * tanhf(c);
        }

        // ---- 64-WG global barrier (per-step counter; release h writes) ----
        __threadfence();
        __syncthreads();
        if (tid == 0) {
            __hip_atomic_fetch_add(&cnt[t], 1u, __ATOMIC_RELEASE, __HIP_MEMORY_SCOPE_AGENT);
            while (__hip_atomic_load(&cnt[t], __ATOMIC_ACQUIRE, __HIP_MEMORY_SCOPE_AGENT) < NWG)
                __builtin_amdgcn_s_sleep(2);
        }
        __syncthreads();
        __threadfence();   // acquire: invalidate L0 before reading new h
    }
}

// relu(h @ W1.T + b1) @ W2.T + b2 -> scalar
__global__ __launch_bounds__(1024, 1)
void lstm_head(const float* __restrict__ hfin,
               const float* __restrict__ W1, const float* __restrict__ b1,
               const float* __restrict__ W2, const float* __restrict__ b2,
               float* __restrict__ out)
{
    __shared__ float h_lds[HID];
    __shared__ float wsum[32];
    const int tid = threadIdx.x;
    const int lane = tid & 31, wave = tid >> 5;
    h_lds[tid] = hfin[tid];
    __syncthreads();
    float acc = 0.f;
    for (int r0 = 0; r0 < 32; ++r0) {
        const int r = wave * 32 + r0;
        const float* wr = W1 + (size_t)r * HID;
        float p = 0.f;
        #pragma unroll 8
        for (int c = 0; c < 32; ++c)
            p = __builtin_fmaf(wr[c * 32 + lane], h_lds[c * 32 + lane], p);
        #pragma unroll
        for (int s = 16; s > 0; s >>= 1) p += __shfl_xor(p, s, 32);
        if (lane == 0) {
            const float z = fmaxf(p + b1[r], 0.f);
            acc = __builtin_fmaf(z, W2[r], acc);
        }
    }
    if (lane == 0) wsum[wave] = acc;
    __syncthreads();
    if (tid == 0) {
        float s = 0.f;
        for (int i = 0; i < 32; ++i) s += wsum[i];
        out[0] = s + b2[0];
    }
}

extern "C" void kernel_launch(void* const* d_in, const int* in_sizes, int n_in,
                              void* d_out, int out_size, void* d_ws, size_t ws_size,
                              hipStream_t stream) {
    (void)in_sizes; (void)n_in; (void)out_size; (void)ws_size;
    const float* x    = (const float*)d_in[0];
    const float* W_ih = (const float*)d_in[1];
    const float* W_hh = (const float*)d_in[2];
    const float* b_ih = (const float*)d_in[3];
    const float* b_hh = (const float*)d_in[4];
    const float* W1   = (const float*)d_in[5];
    const float* b1   = (const float*)d_in[6];
    const float* W2   = (const float*)d_in[7];
    const float* b2   = (const float*)d_in[8];

    char* ws = (char*)d_ws;
    unsigned* cnt = (unsigned*)ws;                    // 16384*4 = 65536 B
    float* hbuf   = (float*)(ws + 65536);             // 2*1024*4 = 8192 B
    float* xg     = (float*)(ws + 65536 + 8192);      // 64*512*64*4 = 8 MiB

    // zero barrier counters + h state every launch (graph-capturable memset)
    (void)hipMemsetAsync(ws, 0, 65536 + 8192, stream);

    lstm_persistent<<<NWG, NTHR, 0, stream>>>(x, W_ih, W_hh, b_ih, b_hh, cnt, hbuf, xg);
    // final h is in buffer (T_STEPS & 1) == 0
    lstm_head<<<1, 1024, 0, stream>>>(hbuf, W1, b1, W2, b2, (float*)d_out);
}
